// SequenceGraph_39556648796626
// MI455X (gfx1250) — matrix operand order repaired and checked by reading, not compiled, not run on hardware
//
#include <hip/hip_runtime.h>
#include <hip/hip_bf16.h>
#include <hip/hip_fp16.h>

typedef __attribute__((ext_vector_type(16))) _Float16 v16h;
typedef __attribute__((ext_vector_type(8)))  float    v8f;

#define N_BATCH 16
#define CIN     128
#define LIN     2048
#define COUT    256
#define KW      8
#define S_OUT   2041          // LIN - KW + 1
#define SP      2048          // S padded to multiple of 16
#define KDIM    (CIN * KW)    // 1024 (im2col reduction dim)

static __device__ __forceinline__ int lane_id() { return (int)(threadIdx.x & 31); }

// ---- WMMA fragment loaders (gfx1250 wave32 layouts, CDNA5 ISA) ----

// A fragment 16x32 f16 from row-major [rows][ld]: lanes 0-15 -> M=lane, K groups
// [k0..k0+7] and [k0+16..k0+23]; lanes 16-31 -> groups [k0+8..] and [k0+24..].
static __device__ __forceinline__ v16h load_a_rowmajor(const _Float16* __restrict__ src,
                                                       int rowbase, int ld, int k0) {
  int lane = lane_id();
  int m  = lane & 15;
  int hi = lane >> 4;
  const _Float16* p = src + (size_t)(rowbase + m) * ld + k0 + hi * 8;
  v16h a;
#pragma unroll
  for (int j = 0; j < 8; ++j) a[j] = p[j];
#pragma unroll
  for (int j = 0; j < 8; ++j) a[8 + j] = p[16 + j];
  return a;
}

// B fragment 32x16 f16 from B^T stored row-major [nrows][ld] (N rows, K contiguous):
// lane holds N = nbase + (lane&15); lo half-wave K groups [k0..k0+7] and [k0+16..k0+23],
// hi half-wave K groups [k0+8..k0+15] and [k0+24..k0+31].
static __device__ __forceinline__ v16h load_b_nmajor(const _Float16* __restrict__ bt,
                                                     int nbase, int ld, int k0) {
  int lane = lane_id();
  int n  = lane & 15;
  int hi = lane >> 4;
  const _Float16* p = bt + (size_t)(nbase + n) * ld + k0 + hi * 8;
  v16h b;
#pragma unroll
  for (int j = 0; j < 8; ++j) b[j] = p[j];
#pragma unroll
  for (int j = 0; j < 8; ++j) b[8 + j] = p[16 + j];
  return b;
}

// ---------------------------------------------------------------------------
// f32 -> f16 conversion
__global__ void cvt_f32_f16_kernel(const float* __restrict__ src,
                                   _Float16* __restrict__ dst, int n) {
  int i = blockIdx.x * blockDim.x + threadIdx.x;
  if (i < n) dst[i] = (_Float16)src[i];
}

// ---------------------------------------------------------------------------
// Conv1d (valid) as im2col WMMA GEMM + bias + ReLU -> h16 [N*SP][COUT] f16.
// One wave per 16(row) x 16(cout) tile, K loop over 1024 in steps of 32.
__global__ void conv_gemm_kernel(const _Float16* __restrict__ x16,
                                 const _Float16* __restrict__ wp16,
                                 const float* __restrict__ conv_b,
                                 _Float16* __restrict__ h16) {
  int wid = (int)((blockIdx.x * blockDim.x + threadIdx.x) >> 5);
  int rowtile = wid >> 4;          // 0 .. N*SP/16-1
  int ct      = wid & 15;          // cout tile
  int rowbase = rowtile * 16;

  int lane = lane_id();
  int m  = lane & 15;
  int hi = lane >> 4;

  int row = rowbase + m;
  int n   = row >> 11;             // / SP
  int s   = row & (SP - 1);
  int se  = s < S_OUT ? s : (S_OUT - 1);   // clamp so pad rows stay in-bounds
  const _Float16* xrow = x16 + (size_t)n * CIN * LIN + se;

  v8f acc;
#pragma unroll
  for (int e = 0; e < 8; ++e) acc[e] = 0.0f;

  for (int k0 = 0; k0 < KDIM; k0 += 32) {
    // A: im2col on the fly. Flattened K index = ci*KW + kk; each 8-wide group
    // is one ci with kk=0..7, i.e. 8 contiguous halves of x16.
    int b1  = k0 + hi * 8;
    int ci1 = b1 >> 3;
    int ci2 = (b1 + 16) >> 3;
    const _Float16* p1 = xrow + (size_t)ci1 * LIN;
    const _Float16* p2 = xrow + (size_t)ci2 * LIN;
    v16h a;
#pragma unroll
    for (int j = 0; j < 8; ++j) a[j] = p1[j];
#pragma unroll
    for (int j = 0; j < 8; ++j) a[8 + j] = p2[j];

    v16h b = load_b_nmajor(wp16, ct * 16, KDIM, k0);
    acc = __builtin_amdgcn_wmma_f32_16x16x32_f16(false, a, false, b,
                                                 (short)0, acc, false, false);
  }

  int c = ct * 16 + (lane & 15);
  float bias = conv_b[c];
#pragma unroll
  for (int r = 0; r < 8; ++r) {
    int rrow = rowbase + hi * 8 + r;
    int ss   = rrow & (SP - 1);
    float vv = acc[r] + bias;
    vv = vv > 0.0f ? vv : 0.0f;
    h16[(size_t)rrow * COUT + c] = (ss < S_OUT) ? (_Float16)vv : (_Float16)0.0f;
  }
}

// ---------------------------------------------------------------------------
// Linear y = h @ W^T + b -> f16, row-major [N*SP][COUT] (pad rows zeroed).
__global__ void linear_kernel(const _Float16* __restrict__ h16,
                              const _Float16* __restrict__ w16,
                              const float* __restrict__ bias,
                              float scale,
                              _Float16* __restrict__ dst) {
  int wid = (int)((blockIdx.x * blockDim.x + threadIdx.x) >> 5);
  int rowtile = wid >> 4;
  int ct      = wid & 15;
  int rowbase = rowtile * 16;

  v8f acc;
#pragma unroll
  for (int e = 0; e < 8; ++e) acc[e] = 0.0f;

#pragma unroll
  for (int k0 = 0; k0 < COUT; k0 += 32) {
    v16h a = load_a_rowmajor(h16, rowbase, COUT, k0);
    v16h b = load_b_nmajor(w16, ct * 16, COUT, k0);
    acc = __builtin_amdgcn_wmma_f32_16x16x32_f16(false, a, false, b,
                                                 (short)0, acc, false, false);
  }

  int lane = lane_id();
  int hi = lane >> 4;
  int d  = ct * 16 + (lane & 15);
  float bb = bias[d];
#pragma unroll
  for (int r = 0; r < 8; ++r) {
    int rrow = rowbase + hi * 8 + r;
    int ss   = rrow & (SP - 1);
    float vv = (acc[r] + bb) * scale;
    dst[(size_t)rrow * COUT + d] = (ss < S_OUT) ? (_Float16)vv : (_Float16)0.0f;
  }
}

// Same linear, but stores TRANSPOSED per batch: dstT[n][d][t] (t contiguous).
// This makes V fragments for P·V loadable as two runs of 8 contiguous halves per lane.
__global__ void linear_t_kernel(const _Float16* __restrict__ h16,
                                const _Float16* __restrict__ w16,
                                const float* __restrict__ bias,
                                _Float16* __restrict__ dstT) {
  int wid = (int)((blockIdx.x * blockDim.x + threadIdx.x) >> 5);
  int rowtile = wid >> 4;
  int ct      = wid & 15;
  int rowbase = rowtile * 16;

  v8f acc;
#pragma unroll
  for (int e = 0; e < 8; ++e) acc[e] = 0.0f;

#pragma unroll
  for (int k0 = 0; k0 < COUT; k0 += 32) {
    v16h a = load_a_rowmajor(h16, rowbase, COUT, k0);
    v16h b = load_b_nmajor(w16, ct * 16, COUT, k0);
    acc = __builtin_amdgcn_wmma_f32_16x16x32_f16(false, a, false, b,
                                                 (short)0, acc, false, false);
  }

  int lane = lane_id();
  int hi = lane >> 4;
  int d  = ct * 16 + (lane & 15);
  float bb = bias[d];
#pragma unroll
  for (int r = 0; r < 8; ++r) {
    int rrow = rowbase + hi * 8 + r;
    int n    = rrow >> 11;
    int ss   = rrow & (SP - 1);
    float vv = acc[r] + bb;
    dstT[((size_t)n * COUT + d) * SP + ss] = (ss < S_OUT) ? (_Float16)vv : (_Float16)0.0f;
  }
}

// ---------------------------------------------------------------------------
// Sigmoid attention: out[n][d][s] = sum_t sigmoid(q[s]·k[t]) * v[t][d].
// One wave per (n, 32-row s-tile pair): every K fragment feeds 2 score tiles,
// every V fragment feeds 2 P·V tiles (2x L2-traffic reuse). Q tiles live in
// registers; P bounces through a per-wave LDS slice (C/D layout -> A layout;
// same-wave LDS ops are in-order).
__global__ void attn_kernel(const _Float16* __restrict__ q16,
                            const _Float16* __restrict__ k16,
                            const _Float16* __restrict__ vT,
                            float* __restrict__ out) {
  __shared__ _Float16 plds[4 * 2 * 16 * 32];   // 4 waves/block, 2 KB each

  int widb = (int)(threadIdx.x >> 5);
  int wid  = (int)((blockIdx.x * blockDim.x + threadIdx.x) >> 5);
  int n  = wid >> 6;          // / (SP/32)
  int sp = wid & 63;          // 32-row s-tile pair index
  int rowbase0 = n * SP + sp * 32;

  int lane = lane_id();
  int m16 = lane & 15;
  int hi  = lane >> 4;
  _Float16* my0 = plds + widb * (2 * 16 * 32);
  _Float16* my1 = my0 + 16 * 32;

  const _Float16* kn  = k16 + (size_t)n * SP * COUT;   // [t][d]
  const _Float16* vTn = vT  + (size_t)n * COUT * SP;   // [d][t]

  v16h qf0[8], qf1[8];
#pragma unroll
  for (int d0 = 0; d0 < 8; ++d0) {
    qf0[d0] = load_a_rowmajor(q16, rowbase0,      COUT, d0 * 32);
    qf1[d0] = load_a_rowmajor(q16, rowbase0 + 16, COUT, d0 * 32);
  }

  v8f acc0[16], acc1[16];
#pragma unroll
  for (int i = 0; i < 16; ++i)
#pragma unroll
    for (int e = 0; e < 8; ++e) { acc0[i][e] = 0.0f; acc1[i][e] = 0.0f; }

  for (int tb = 0; tb < SP; tb += 32) {
    // prefetch next chunk toward the WGP while this chunk computes
    if (tb + 32 < SP) {
      __builtin_prefetch(kn + (size_t)(tb + 32) * COUT, 0, 0);
      __builtin_prefetch(vTn + (tb + 32), 0, 0);
    }

    v8f s00, s01, s10, s11;   // [stile][t-half]
#pragma unroll
    for (int e = 0; e < 8; ++e) { s00[e] = s01[e] = s10[e] = s11[e] = 0.0f; }

#pragma unroll
    for (int d0 = 0; d0 < 8; ++d0) {
      v16h b0 = load_b_nmajor(kn, tb,      COUT, d0 * 32);
      v16h b1 = load_b_nmajor(kn, tb + 16, COUT, d0 * 32);
      s00 = __builtin_amdgcn_wmma_f32_16x16x32_f16(false, qf0[d0], false, b0,
                                                   (short)0, s00, false, false);
      s01 = __builtin_amdgcn_wmma_f32_16x16x32_f16(false, qf0[d0], false, b1,
                                                   (short)0, s01, false, false);
      s10 = __builtin_amdgcn_wmma_f32_16x16x32_f16(false, qf1[d0], false, b0,
                                                   (short)0, s10, false, false);
      s11 = __builtin_amdgcn_wmma_f32_16x16x32_f16(false, qf1[d0], false, b1,
                                                   (short)0, s11, false, false);
    }

    // sigmoid, stage both P tiles (16 s-rows x 32 t-cols each) into LDS
#pragma unroll
    for (int r = 0; r < 8; ++r) {
      int mm = hi * 8 + r;
      my0[mm * 32 + m16]      = (_Float16)(1.0f / (1.0f + __expf(-s00[r])));
      my0[mm * 32 + 16 + m16] = (_Float16)(1.0f / (1.0f + __expf(-s01[r])));
      my1[mm * 32 + m16]      = (_Float16)(1.0f / (1.0f + __expf(-s10[r])));
      my1[mm * 32 + 16 + m16] = (_Float16)(1.0f / (1.0f + __expf(-s11[r])));
    }
    // reload as A fragments (wave-private slice; LDS ops in-order per wave)
    v16h pf0, pf1;
    {
      const _Float16* p0 = my0 + m16 * 32 + hi * 8;
      const _Float16* p1 = my1 + m16 * 32 + hi * 8;
#pragma unroll
      for (int j = 0; j < 8; ++j) { pf0[j] = p0[j];      pf1[j] = p1[j]; }
#pragma unroll
      for (int j = 0; j < 8; ++j) { pf0[8 + j] = p0[16 + j]; pf1[8 + j] = p1[16 + j]; }
    }

    // P·V: V fragment (32 t x 16 d) from vT[d][t] = contiguous halves per lane
#pragma unroll
    for (int dt = 0; dt < 16; ++dt) {
      v16h vb = load_b_nmajor(vTn, dt * 16, SP, tb);
      acc0[dt] = __builtin_amdgcn_wmma_f32_16x16x32_f16(false, pf0, false, vb,
                                                        (short)0, acc0[dt], false, false);
      acc1[dt] = __builtin_amdgcn_wmma_f32_16x16x32_f16(false, pf1, false, vb,
                                                        (short)0, acc1[dt], false, false);
    }
  }

  // store transposed to [N][COUT][S_OUT]
#pragma unroll
  for (int dt = 0; dt < 16; ++dt) {
    int d = dt * 16 + m16;
    float* od = out + ((size_t)n * COUT + d) * S_OUT;
#pragma unroll
    for (int r = 0; r < 8; ++r) {
      int s0i = sp * 32 + hi * 8 + r;
      if (s0i < S_OUT) od[s0i] = acc0[dt][r];
      int s1i = s0i + 16;
      if (s1i < S_OUT) od[s1i] = acc1[dt][r];
    }
  }
}

// ---------------------------------------------------------------------------
extern "C" void kernel_launch(void* const* d_in, const int* in_sizes, int n_in,
                              void* d_out, int out_size, void* d_ws, size_t ws_size,
                              hipStream_t stream) {
  const float* x      = (const float*)d_in[0];
  const float* conv_w = (const float*)d_in[1];
  const float* conv_b = (const float*)d_in[2];
  const float* wq     = (const float*)d_in[3];
  const float* bq     = (const float*)d_in[4];
  const float* wk     = (const float*)d_in[5];
  const float* bk     = (const float*)d_in[6];
  const float* wv     = (const float*)d_in[7];
  const float* bv     = (const float*)d_in[8];
  float* out = (float*)d_out;

  // Workspace layout (all f16). q16 aliases x16: x16 is dead before the q
  // projection launches (stream-ordered).
  char* ws = (char*)d_ws;
  size_t off = 0;
  auto take = [&](size_t bytes) -> char* {
    char* p = ws + off;
    off = (off + bytes + 255) & ~(size_t)255;
    return p;
  };
  const size_t rowsz = (size_t)N_BATCH * SP * COUT * sizeof(_Float16); // 16.8 MB
  _Float16* q16  = (_Float16*)take(rowsz);
  _Float16* x16  = (_Float16*)q16;                       // alias (dead before q)
  _Float16* k16  = (_Float16*)take(rowsz);
  _Float16* vT16 = (_Float16*)take(rowsz);
  _Float16* h16  = (_Float16*)take(rowsz);
  _Float16* wp16 = (_Float16*)take((size_t)COUT * KDIM * sizeof(_Float16));
  _Float16* wq16 = (_Float16*)take((size_t)COUT * COUT * sizeof(_Float16));
  _Float16* wk16 = (_Float16*)take((size_t)COUT * COUT * sizeof(_Float16));
  _Float16* wv16 = (_Float16*)take((size_t)COUT * COUT * sizeof(_Float16));

  // 1) convert inputs to f16
  {
    int nx = N_BATCH * CIN * LIN;
    cvt_f32_f16_kernel<<<(nx + 255) / 256, 256, 0, stream>>>(x, x16, nx);
    int nw = COUT * KDIM;
    cvt_f32_f16_kernel<<<(nw + 255) / 256, 256, 0, stream>>>(conv_w, wp16, nw);
    int nl = COUT * COUT;
    cvt_f32_f16_kernel<<<(nl + 255) / 256, 256, 0, stream>>>(wq, wq16, nl);
    cvt_f32_f16_kernel<<<(nl + 255) / 256, 256, 0, stream>>>(wk, wk16, nl);
    cvt_f32_f16_kernel<<<(nl + 255) / 256, 256, 0, stream>>>(wv, wv16, nl);
  }

  // 2) conv as im2col WMMA GEMM: (N*SP/16)*(COUT/16) = 32768 waves
  {
    int waves = (N_BATCH * SP / 16) * (COUT / 16);
    conv_gemm_kernel<<<waves * 32 / 256, 256, 0, stream>>>(x16, wp16, conv_b, h16);
  }

  // 3) Q/K/V projections (q carries the 1/sqrt(C) score scale; V transposed)
  {
    int waves = (N_BATCH * SP / 16) * (COUT / 16);
    int blocks = waves * 32 / 256;
    linear_kernel<<<blocks, 256, 0, stream>>>(h16, wk16, bk, 1.0f, k16);
    linear_t_kernel<<<blocks, 256, 0, stream>>>(h16, wv16, bv, vT16);
    linear_kernel<<<blocks, 256, 0, stream>>>(h16, wq16, bq, 1.0f / 16.0f, q16);
  }

  // 4) sigmoid attention + output transpose: 16*(SP/32) = 1024 waves
  {
    int waves = N_BATCH * (SP / 32);
    attn_kernel<<<waves * 32 / 128, 128, 0, stream>>>(q16, k16, vT16, out);
  }
}
